// alexnet_39968965657132
// MI455X (gfx1250) — 16-bit and 8-bit matrix operand order repaired and checked by reading, compiled, not run on hardware
//
#include <hip/hip_runtime.h>

typedef __attribute__((ext_vector_type(16))) _Float16 v16h;
typedef __attribute__((ext_vector_type(8)))  float    v8f;
typedef __attribute__((ext_vector_type(8)))  int      v8i;

#define EPS 1e-3f
#define LDT 80

__device__ __forceinline__ void async_copy_b128(unsigned lds_off, const void* gptr) {
    asm volatile("global_load_async_to_lds_b128 %0, %1, off"
                 :: "v"(lds_off), "v"((unsigned long long)(size_t)gptr)
                 : "memory");
}
__device__ __forceinline__ void async_copy_b64(unsigned lds_off, const void* gptr) {
    asm volatile("global_load_async_to_lds_b64 %0, %1, off"
                 :: "v"(lds_off), "v"((unsigned long long)(size_t)gptr)
                 : "memory");
}
__device__ __forceinline__ void async_wait_all() {
    asm volatile("s_wait_asynccnt 0x0" ::: "memory");
}

__device__ inline v8i frag_ld_i(const signed char* rowbase, int lane) {
    const int4* p = (const int4*)(rowbase + (lane >> 4) * 32);
    union { v8i v; int4 q[2]; } u;
    u.q[0] = p[0]; u.q[1] = p[1];
    return u.v;
}
__device__ inline v16h frag_ld_h(const char* rowbase, int lane) {
    const int4* p = (const int4*)(rowbase + (lane >> 4) * 32);
    union { v16h v; int4 q[2]; } u;
    u.q[0] = p[0]; u.q[1] = p[1];
    return u.v;
}

__global__ __launch_bounds__(256)
void bconv_iu8_kernel(const signed char* __restrict__ act,
                      const signed char* __restrict__ wb,
                      float* __restrict__ fout, signed char* __restrict__ bout,
                      const float* __restrict__ gamma, const float* __restrict__ beta,
                      const float* __restrict__ mean,  const float* __restrict__ var,
                      int H, int W, int Cin, int Cout, int KW, int pad,
                      int K, int sign_out)
{
    __shared__ __align__(16) signed char As[2][128 * LDT];
    __shared__ __align__(16) signed char Bs[2][64 * LDT];

    const int tid   = threadIdx.x;
    const int lane  = tid & 31;
    const int wave  = tid >> 5;
    const int wm    = wave >> 1;
    const int wn    = wave & 1;
    const int mtile = blockIdx.x;
    const int ntile = blockIdx.y;
    const int HW    = H * W;
    const int Ksteps = K >> 6;

    const int ar   = tid >> 1;
    const int ahal = tid & 1;
    const int am   = mtile * 128 + ar;
    const int ab   = am / HW;
    const int arem = am - ab * HW;
    const int aoh  = arem / W;
    const int aow  = arem - aoh * W;
    const size_t abase = (size_t)ab * H;

    const int bn_ = tid >> 2;
    const int bq  = tid & 3;
    const signed char* bptr = wb + (size_t)(ntile * 64 + bn_) * K + bq * 16;
    const unsigned bswz = (unsigned)((bq & 1) * 32 + ((bq >> 1) & 1) * 16);

    int kh = 0, kw = 0, cs = 0, k0n = 0;

    auto load_tiles = [&](int buf) {
        const int ih = aoh - pad + kh;
        const int iw = aow - pad + kw;
        const unsigned arow = (unsigned)(size_t)&As[buf][ar * LDT];
        if (ih >= 0 && ih < H && iw >= 0 && iw < W) {
            const signed char* src =
                act + ((abase + ih) * W + iw) * Cin + cs + ahal * 32;
#pragma unroll
            for (int p = 0; p < 4; ++p) {
                int j = ahal * 4 + p;
                async_copy_b64(arow + ((j & 1) + ((j >> 1) & 1) * 4 + (j >> 2) * 2) * 8, src + p * 8);
            }
        } else {
            int4 z = make_int4(0, 0, 0, 0);
            *(int4*)&As[buf][ar * LDT + ahal * 16]      = z;
            *(int4*)&As[buf][ar * LDT + 32 + ahal * 16] = z;
        }
        async_copy_b128((unsigned)(size_t)&Bs[buf][bn_ * LDT] + bswz, bptr + k0n);
        k0n += 64;
        cs  += 64;
        if (cs == Cin) { cs = 0; if (++kw == KW) { kw = 0; ++kh; } }
    };

    load_tiles(0);

    v8i acc[2][2] = {};

    for (int kk = 0; kk < Ksteps; ++kk) {
        async_wait_all();
        __syncthreads();
        const int cur = kk & 1;
        if (kk + 1 < Ksteps) load_tiles(cur ^ 1);

        const signed char* at = &As[cur][(wm * 32 + (lane & 15)) * LDT];
        const signed char* bt = &Bs[cur][(wn * 32 + (lane & 15)) * LDT];
        v8i a0 = frag_ld_i(at,            lane);
        v8i a1 = frag_ld_i(at + 16 * LDT, lane);
        v8i b0 = frag_ld_i(bt,            lane);
        v8i b1 = frag_ld_i(bt + 16 * LDT, lane);

        acc[0][0] = __builtin_amdgcn_wmma_i32_16x16x64_iu8(true, a0, true, b0, acc[0][0], false, false);
        acc[0][1] = __builtin_amdgcn_wmma_i32_16x16x64_iu8(true, a0, true, b1, acc[0][1], false, false);
        acc[1][0] = __builtin_amdgcn_wmma_i32_16x16x64_iu8(true, a1, true, b0, acc[1][0], false, false);
        acc[1][1] = __builtin_amdgcn_wmma_i32_16x16x64_iu8(true, a1, true, b1, acc[1][1], false, false);
    }

    const int l16  = lane >> 4;
    const int ncol = lane & 15;
#pragma unroll
    for (int mi = 0; mi < 2; ++mi) {
#pragma unroll
        for (int ni = 0; ni < 2; ++ni) {
            int n  = ntile * 64 + wn * 32 + ni * 16 + ncol;
            float g  = gamma[n], bt = beta[n];
            float mu = mean[n],  iv = rsqrtf(var[n] + EPS);
#pragma unroll
            for (int r = 0; r < 8; ++r) {
                int m = mtile * 128 + wm * 32 + mi * 16 + r + l16 * 8;
                float v = (float)acc[mi][ni][r];
                v = fmaxf(v, 0.0f);
                v = (v - mu) * iv * g + bt;
                size_t o = (size_t)m * Cout + n;
                if (sign_out) bout[o] = (v >= 0.0f) ? (signed char)1 : (signed char)-1;
                else          fout[o] = v;
            }
        }
    }
}

__global__ __launch_bounds__(256)
void fc_f16_kernel(const _Float16* __restrict__ A, const _Float16* __restrict__ Bw,
                   const float* __restrict__ bias,
                   const float* __restrict__ gamma, const float* __restrict__ beta,
                   const float* __restrict__ mean,  const float* __restrict__ var,
                   float* __restrict__ outF, _Float16* __restrict__ outH,
                   int K, int Nout, int ldo, int relu_bn)
{
    __shared__ __align__(16) char Ah[2][64 * LDT];
    __shared__ __align__(16) char Bh[2][64 * LDT];

    const int tid   = threadIdx.x;
    const int lane  = tid & 31;
    const int wave  = tid >> 5;
    const int wm    = wave >> 1;
    const int wn    = wave & 1;
    const int ntile = blockIdx.x;
    const int Ksteps = K >> 5;

    const int lr = tid >> 2;
    const int lq = tid & 3;
    const _Float16* aptr = A  + (size_t)lr * K + lq * 8;
    const _Float16* bptr = Bw + (size_t)(ntile * 64 + lr) * K + lq * 8;
    const unsigned aswz = (unsigned)((lq & 1) * 32 + ((lq >> 1) & 1) * 16);
    const unsigned bswz = (unsigned)((lq & 1) * 32 + ((lq >> 1) & 1) * 16);

    int k0n = 0;
    auto load_tiles = [&](int buf) {
        async_copy_b128((unsigned)(size_t)&Ah[buf][lr * LDT] + aswz, aptr + k0n);
        async_copy_b128((unsigned)(size_t)&Bh[buf][lr * LDT] + bswz, bptr + k0n);
        k0n += 32;
    };

    load_tiles(0);

    v8f acc0 = {}, acc1 = {};

    for (int kk = 0; kk < Ksteps; ++kk) {
        async_wait_all();
        __syncthreads();
        const int cur = kk & 1;
        if (kk + 1 < Ksteps) load_tiles(cur ^ 1);

        const char* at = &Ah[cur][(wm * 16 + (lane & 15)) * LDT];
        const char* bt = &Bh[cur][(wn * 32 + (lane & 15)) * LDT];
        v16h a  = frag_ld_h(at,            lane);
        v16h b0 = frag_ld_h(bt,            lane);
        v16h b1 = frag_ld_h(bt + 16 * LDT, lane);

        acc0 = __builtin_amdgcn_wmma_f32_16x16x32_f16(false, a, false, b0, (short)0, acc0, false, false);
        acc1 = __builtin_amdgcn_wmma_f32_16x16x32_f16(false, a, false, b1, (short)0, acc1, false, false);
    }

    const int l16  = lane >> 4;
    const int ncol = lane & 15;
#pragma unroll
    for (int ni = 0; ni < 2; ++ni) {
        int n = ntile * 64 + wn * 32 + ni * 16 + ncol;
        if (n >= Nout) continue;
        float bs = bias[n];
        float g = 1.0f, bt = 0.0f, mu = 0.0f, iv = 1.0f;
        if (relu_bn) { g = gamma[n]; bt = beta[n]; mu = mean[n]; iv = rsqrtf(var[n] + EPS); }
        v8f acc = (ni == 0) ? acc0 : acc1;
#pragma unroll
        for (int r = 0; r < 8; ++r) {
            int m = wm * 16 + r + l16 * 8;
            float v = acc[r] + bs;
            if (relu_bn) { v = fmaxf(v, 0.0f); v = (v - mu) * iv * g + bt; }
            size_t o = (size_t)m * ldo + n;
            if (outH) outH[o] = (_Float16)v;
            else      outF[o] = v;
        }
    }
}

__global__ __launch_bounds__(256)
void conv1_fused_kernel(const float* __restrict__ x, const float* __restrict__ w1,
                        const float* __restrict__ b1,
                        const float* __restrict__ mean, const float* __restrict__ var,
                        signed char* __restrict__ out, int total)
{
    int idx = blockIdx.x * 256 + threadIdx.x;
    if (idx >= total) return;
    int c = idx & 63;
    int t = idx >> 6;
    int pw = t % 28; t /= 28;
    int ph = t % 28;
    int b  = t / 28;

    float mu = mean[c];
    float iv = rsqrtf(var[c] + EPS);
    float bias = b1[c];
    float mx = -1e30f;

#pragma unroll
    for (int dy = 0; dy < 2; ++dy)
#pragma unroll
    for (int dx = 0; dx < 2; ++dx) {
        int oh = ph * 2 + dy, ow = pw * 2 + dx;
        float s = bias;
        for (int kh = 0; kh < 11; ++kh) {
            int ih = oh * 4 - 3 + kh;
            if (ih < 0 || ih >= 224) continue;
            for (int kw = 0; kw < 11; ++kw) {
                int iw = ow * 4 - 3 + kw;
                if (iw < 0 || iw >= 224) continue;
                const float* xp = x  + (((size_t)b * 224 + ih) * 224 + iw) * 3;
                const float* wp = w1 + ((size_t)(kh * 11 + kw) * 3) * 64 + c;
                s = fmaf(xp[0], wp[0],   s);
                s = fmaf(xp[1], wp[64],  s);
                s = fmaf(xp[2], wp[128], s);
            }
        }
        s = fmaxf(s, 0.0f);
        s = (s - mu) * iv;
        mx = fmaxf(mx, s);
    }
    out[idx] = (mx >= 0.0f) ? (signed char)1 : (signed char)-1;
}

__global__ __launch_bounds__(256)
void maxpool2_bin_kernel(const float* __restrict__ in, signed char* __restrict__ out,
                         int OH, int OW, int C, int total)
{
    int idx = blockIdx.x * 256 + threadIdx.x;
    if (idx >= total) return;
    int c = idx % C; int t = idx / C;
    int ow = t % OW; t /= OW;
    int oh = t % OH; int b = t / OH;
    int H = OH * 2, W = OW * 2;
    const float* p = in + (((size_t)b * H + oh * 2) * W + ow * 2) * C + c;
    size_t rs = (size_t)W * C;
    float m = fmaxf(fmaxf(p[0], p[C]), fmaxf(p[rs], p[rs + C]));
    out[idx] = (m >= 0.0f) ? (signed char)1 : (signed char)-1;
}

__global__ __launch_bounds__(256)
void maxpool2_f16_kernel(const float* __restrict__ in, _Float16* __restrict__ out,
                         int OH, int OW, int C, int total)
{
    int idx = blockIdx.x * 256 + threadIdx.x;
    if (idx >= total) return;
    int c = idx % C; int t = idx / C;
    int ow = t % OW; t /= OW;
    int oh = t % OH; int b = t / OH;
    int H = OH * 2, W = OW * 2;
    const float* p = in + (((size_t)b * H + oh * 2) * W + ow * 2) * C + c;
    size_t rs = (size_t)W * C;
    float m = fmaxf(fmaxf(p[0], p[C]), fmaxf(p[rs], p[rs + C]));
    out[idx] = (_Float16)m;
}

__global__ __launch_bounds__(256)
void binw_kernel(const float* __restrict__ w, signed char* __restrict__ wb,
                 int K, int Cout, int total)
{
    int idx = blockIdx.x * 256 + threadIdx.x;
    if (idx >= total) return;
    int co = idx / K;
    int k  = idx - co * K;
    wb[idx] = (w[(size_t)k * Cout + co] >= 0.0f) ? (signed char)1 : (signed char)-1;
}

__global__ __launch_bounds__(256)
void trw_f16_kernel(const float* __restrict__ w, _Float16* __restrict__ wt,
                    int K, int N, int total)
{
    int idx = blockIdx.x * 256 + threadIdx.x;
    if (idx >= total) return;
    int n = idx / K;
    int k = idx - n * K;
    float v = (n < N) ? w[(size_t)k * N + n] : 0.0f;
    wt[idx] = (_Float16)v;
}

__global__ __launch_bounds__(256)
void softmax_kernel(const float* __restrict__ logits, float* __restrict__ out,
                    int N, int ld)
{
    __shared__ float red[256];
    int b = blockIdx.x, t = threadIdx.x;
    const float* row = logits + (size_t)b * ld;

    float mx = -1e30f;
    for (int i = t; i < N; i += 256) mx = fmaxf(mx, row[i]);
    red[t] = mx; __syncthreads();
    for (int s = 128; s > 0; s >>= 1) { if (t < s) red[t] = fmaxf(red[t], red[t + s]); __syncthreads(); }
    mx = red[0]; __syncthreads();

    float sum = 0.0f;
    for (int i = t; i < N; i += 256) sum += __expf(row[i] - mx);
    red[t] = sum; __syncthreads();
    for (int s = 128; s > 0; s >>= 1) { if (t < s) red[t] += red[t + s]; __syncthreads(); }
    sum = red[0];

    float inv = 1.0f / sum;
    for (int i = t; i < N; i += 256)
        out[(size_t)b * N + i] = __expf(row[i] - mx) * inv;
}

extern "C" void kernel_launch(void* const* d_in, const int* in_sizes, int n_in,
                              void* d_out, int out_size, void* d_ws, size_t ws_size,
                              hipStream_t stream) {
    (void)in_sizes; (void)n_in; (void)out_size; (void)ws_size;

    const float* x        = (const float*)d_in[0];
    const float* w1       = (const float*)d_in[1];
    const float* b1       = (const float*)d_in[2];
    const float* bn1_mean = (const float*)d_in[3];
    const float* bn1_var  = (const float*)d_in[4];
    const float* w2       = (const float*)d_in[5];
    const float* bn2_g    = (const float*)d_in[6];
    const float* bn2_b    = (const float*)d_in[7];
    const float* bn2_m    = (const float*)d_in[8];
    const float* bn2_v    = (const float*)d_in[9];
    const float* w3       = (const float*)d_in[10];
    const float* bn3_g    = (const float*)d_in[11];
    const float* bn3_b    = (const float*)d_in[12];
    const float* bn3_m    = (const float*)d_in[13];
    const float* bn3_v    = (const float*)d_in[14];
    const float* w4       = (const float*)d_in[15];
    const float* bn4_g    = (const float*)d_in[16];
    const float* bn4_b    = (const float*)d_in[17];
    const float* bn4_m    = (const float*)d_in[18];
    const float* bn4_v    = (const float*)d_in[19];
    const float* w5       = (const float*)d_in[20];
    const float* bn5_g    = (const float*)d_in[21];
    const float* bn5_b    = (const float*)d_in[22];
    const float* bn5_m    = (const float*)d_in[23];
    const float* bn5_v    = (const float*)d_in[24];
    const float* w6       = (const float*)d_in[25];
    const float* b6       = (const float*)d_in[26];
    const float* bn6_g    = (const float*)d_in[27];
    const float* bn6_b    = (const float*)d_in[28];
    const float* bn6_m    = (const float*)d_in[29];
    const float* bn6_v    = (const float*)d_in[30];
    const float* w7       = (const float*)d_in[31];
    const float* b7       = (const float*)d_in[32];
    const float* bn7_g    = (const float*)d_in[33];
    const float* bn7_b    = (const float*)d_in[34];
    const float* bn7_m    = (const float*)d_in[35];
    const float* bn7_v    = (const float*)d_in[36];
    const float* w8       = (const float*)d_in[37];
    const float* b8       = (const float*)d_in[38];

    const int B = 64;
    const int K2 = 5 * 5 * 64;
    const int K3 = 3 * 3 * 192;
    const int K4 = 3 * 3 * 384;
    const int K5 = 3 * 3 * 384;
    const int K6 = 12544, K7 = 4096, K8 = 4096;
    const int N8pad = 1024;

    char* ws = (char*)d_ws;
    size_t off = 0;
    auto give = [&](size_t bytes) -> size_t {
        size_t o = off;
        off += (bytes + 255) & ~(size_t)255;
        return o;
    };

    signed char* wb2   = (signed char*)(ws + give((size_t)192 * K2));
    signed char* wb3   = (signed char*)(ws + give((size_t)384 * K3));
    signed char* wb4   = (signed char*)(ws + give((size_t)384 * K4));
    signed char* wb5   = (signed char*)(ws + give((size_t)256 * K5));
    _Float16*    w6T   = (_Float16*)  (ws + give((size_t)4096  * K6 * 2));
    _Float16*    w7T   = (_Float16*)  (ws + give((size_t)4096  * K7 * 2));
    _Float16*    w8T   = (_Float16*)  (ws + give((size_t)N8pad * K8 * 2));
    signed char* act1  = (signed char*)(ws + give((size_t)B * 28 * 28 * 64));
    float*       c2out = (float*)     (ws + give((size_t)B * 28 * 28 * 192 * 4));
    signed char* act2  = (signed char*)(ws + give((size_t)B * 14 * 14 * 192));
    signed char* act3  = (signed char*)(ws + give((size_t)B * 14 * 14 * 384));
    signed char* act4  = (signed char*)(ws + give((size_t)B * 14 * 14 * 384));
    float*       c5out = (float*)     (ws + give((size_t)B * 14 * 14 * 256 * 4));
    _Float16*    pool5 = (_Float16*)  (ws + give((size_t)B * 12544 * 2));
    _Float16*    fc6o  = (_Float16*)  (ws + give((size_t)B * 4096 * 2));
    _Float16*    fc7o  = (_Float16*)  (ws + give((size_t)B * 4096 * 2));
    float*       logit = (float*)     (ws + give((size_t)B * N8pad * 4));

    auto blks = [](long long n) { return (unsigned)((n + 255) / 256); };

    binw_kernel<<<blks((long long)192 * K2), 256, 0, stream>>>(w2, wb2, K2, 192, 192 * K2);
    binw_kernel<<<blks((long long)384 * K3), 256, 0, stream>>>(w3, wb3, K3, 384, 384 * K3);
    binw_kernel<<<blks((long long)384 * K4), 256, 0, stream>>>(w4, wb4, K4, 384, 384 * K4);
    binw_kernel<<<blks((long long)256 * K5), 256, 0, stream>>>(w5, wb5, K5, 256, 256 * K5);
    trw_f16_kernel<<<blks((long long)4096  * K6), 256, 0, stream>>>(w6, w6T, K6, 4096, 4096 * K6);
    trw_f16_kernel<<<blks((long long)4096  * K7), 256, 0, stream>>>(w7, w7T, K7, 4096, 4096 * K7);
    trw_f16_kernel<<<blks((long long)N8pad * K8), 256, 0, stream>>>(w8, w8T, K8, 1000, N8pad * K8);

    {
        int total = B * 28 * 28 * 64;
        conv1_fused_kernel<<<blks(total), 256, 0, stream>>>(x, w1, b1, bn1_mean, bn1_var, act1, total);
    }

    {
        dim3 grid((B * 28 * 28) / 128, 192 / 64);
        bconv_iu8_kernel<<<grid, 256, 0, stream>>>(act1, wb2, c2out, nullptr,
            bn2_g, bn2_b, bn2_m, bn2_v, 28, 28, 64, 192, 5, 2, K2, 0);
    }
    {
        int total = B * 14 * 14 * 192;
        maxpool2_bin_kernel<<<blks(total), 256, 0, stream>>>(c2out, act2, 14, 14, 192, total);
    }

    {
        dim3 grid((B * 14 * 14) / 128, 384 / 64);
        bconv_iu8_kernel<<<grid, 256, 0, stream>>>(act2, wb3, nullptr, act3,
            bn3_g, bn3_b, bn3_m, bn3_v, 14, 14, 192, 384, 3, 1, K3, 1);
    }
    {
        dim3 grid((B * 14 * 14) / 128, 384 / 64);
        bconv_iu8_kernel<<<grid, 256, 0, stream>>>(act3, wb4, nullptr, act4,
            bn4_g, bn4_b, bn4_m, bn4_v, 14, 14, 384, 384, 3, 1, K4, 1);
    }
    {
        dim3 grid((B * 14 * 14) / 128, 256 / 64);
        bconv_iu8_kernel<<<grid, 256, 0, stream>>>(act4, wb5, c5out, nullptr,
            bn5_g, bn5_b, bn5_m, bn5_v, 14, 14, 384, 256, 3, 1, K5, 0);
    }
    {
        int total = B * 7 * 7 * 256;
        maxpool2_f16_kernel<<<blks(total), 256, 0, stream>>>(c5out, pool5, 7, 7, 256, total);
    }

    fc_f16_kernel<<<4096 / 64, 256, 0, stream>>>(pool5, w6T, b6,
        bn6_g, bn6_b, bn6_m, bn6_v, nullptr, fc6o, K6, 4096, 4096, 1);
    fc_f16_kernel<<<4096 / 64, 256, 0, stream>>>(fc6o, w7T, b7,
        bn7_g, bn7_b, bn7_m, bn7_v, nullptr, fc7o, K7, 4096, 4096, 1);
    fc_f16_kernel<<<N8pad / 64, 256, 0, stream>>>(fc7o, w8T, b8,
        nullptr, nullptr, nullptr, nullptr, logit, nullptr, K8, 1000, N8pad, 0);

    softmax_kernel<<<B, 256, 0, stream>>>(logit, (float*)d_out, 1000, N8pad);
}
